// CausalSelfAttention_86861418594597
// MI455X (gfx1250) — matrix operand order repaired and checked by reading, not compiled, not run on hardware
//
#include <hip/hip_runtime.h>

// ---------------------------------------------------------------------------
// CDNA5 (gfx1250, wave32) causal self-attention forward.
//   x:[4,2048,1024] f32, Wq/Wk/Wv/Wo:[1024,1024] f32, cos/sin:[2048,64] f32
// Pipeline: cast->bf16, WMMA GEMMs for QKV, RoPE, flash-attention (WMMA),
// WMMA GEMM for output projection (f32 out).
//  - All matmuls: v_wmma_f32_16x16x32_bf16, fp32 accumulation.
//  - GEMM A-tiles staged via TDM tensor_load_to_lds, double-buffered:
//    DMA for tile k+1 overlaps WMMA compute of tile k (s_wait_tensorcnt
//    drains just before the barrier).
//  - GEMM B-tiles / flash V-tiles software-pipelined through registers.
//  - Softmax row reductions via DPP row_ror rotations (no ds_bpermute).
// ---------------------------------------------------------------------------

#ifndef USE_TDM
#define USE_TDM 1
#endif

typedef __attribute__((ext_vector_type(16))) __bf16 v16bf;
typedef __attribute__((ext_vector_type(8)))  float  v8f;
typedef __attribute__((ext_vector_type(4)))  unsigned int v4u;
typedef __attribute__((ext_vector_type(8)))  int    v8i_t;
typedef __attribute__((ext_vector_type(4)))  int    v4i_t;

union FragAB {
  v16bf v;
  uint4 q[2];
  unsigned short s[16];
};

__device__ __forceinline__ unsigned short f2bf(float f) {
  unsigned int u = __float_as_uint(f);
  unsigned int r = u + 0x7FFFu + ((u >> 16) & 1u);   // round-to-nearest-even
  return (unsigned short)(r >> 16);
}
__device__ __forceinline__ float bf2f(unsigned short h) {
  return __uint_as_float(((unsigned int)h) << 16);
}
__device__ __forceinline__ v8f v8f_zero() {
  v8f z = {0.f, 0.f, 0.f, 0.f, 0.f, 0.f, 0.f, 0.f};
  return z;
}

// ---- 16-lane (DPP row) reductions via row_ror rotations -------------------
template <int CTRL>
__device__ __forceinline__ float dpp_rot_f32(float x) {
  return __int_as_float(
      __builtin_amdgcn_update_dpp(0, __float_as_int(x), CTRL, 0xF, 0xF, true));
}
__device__ __forceinline__ float row16_max(float x) {
  x = fmaxf(x, dpp_rot_f32<0x121>(x));  // row_ror:1
  x = fmaxf(x, dpp_rot_f32<0x122>(x));  // row_ror:2
  x = fmaxf(x, dpp_rot_f32<0x124>(x));  // row_ror:4
  x = fmaxf(x, dpp_rot_f32<0x128>(x));  // row_ror:8
  return x;
}
__device__ __forceinline__ float row16_sum(float x) {
  x += dpp_rot_f32<0x121>(x);
  x += dpp_rot_f32<0x122>(x);
  x += dpp_rot_f32<0x124>(x);
  x += dpp_rot_f32<0x128>(x);
  return x;
}

// ------------------------------- f32 -> bf16 -------------------------------
__global__ void cvt_f32_bf16(const float* __restrict__ in,
                             unsigned short* __restrict__ out, int n) {
  int i = blockIdx.x * blockDim.x + threadIdx.x;
  if (i < n) out[i] = f2bf(in[i]);
}

// ------------------------------ WMMA GEMM ----------------------------------
// C[M,N] = A[M,K] * B[K,N], A/B bf16 row-major, C fp32 or bf16.
// Block tile 128x128, 256 threads = 8 waves (4 M-subtiles x 2 N-subtiles),
// each wave computes 32x64 via 2x4 wmma accumulators.
// Double-buffered LDS laid out as [As0 | As1 | Bs0 | Bs1]; buffers addressed
// with offset arithmetic only (no arrays of LDS pointers).
template <bool OUT_F32>
__global__ __launch_bounds__(256) void gemm_bf16_wmma(
    const unsigned short* __restrict__ A,
    const unsigned short* __restrict__ B,
    void* __restrict__ Cout, int M, int N, int K) {
  constexpr int LDT   = 40;          // padded LDS stride: 80B rows
  constexpr int TILE  = 128 * LDT;   // one tile in halves (5120)
  constexpr unsigned TILE_B = TILE * 2u;  // one tile in bytes (10240)
  extern __shared__ __align__(16) unsigned short smem[];
  // layout: smem[0..TILE) = As0, [TILE..2T) = As1, [2T..3T) = Bs0, [3T..4T)=Bs1

  const int tid   = threadIdx.x;
  const int wave  = tid >> 5;
  const int lane  = tid & 31;
  const int l16   = lane & 15;
  const int lhalf = lane >> 4;

  const int wm   = wave & 3;   // 32-row subtile
  const int wn   = wave >> 2;  // 64-col subtile
  const int row0 = blockIdx.y * 128;
  const int col0 = blockIdx.x * 128;

  // ---- helpers --------------------------------------------------------
#if USE_TDM
  // D# per ISA 8.3/8.4: 2-D tensor, data_size=2B, pad_enable with
  // pad_interval=16 DWORDs (one 32-elem line) and pad_amount=4 DWORDs
  // -> hardware produces the padded LDS stride of 40 halves.
  auto tdm_issue = [&](int kt, unsigned ldsoff) {
    unsigned long long ga =
        (unsigned long long)(size_t)(&A[(size_t)row0 * K + kt]);
    v4u g0;
    g0.x = 1u;                                   // count=1 valid descriptor
    g0.y = ldsoff;                               // LDS byte address
    g0.z = (unsigned)(ga & 0xFFFFFFFFu);         // global_addr[31:0]
    g0.w = (unsigned)((ga >> 32) & 0x1FFFFFFu)   // global_addr[56:32]
           | (2u << 30);                         // type=2 ("image")
    v8i_t g1;
    g1[0] = (int)((1u << 16)      // data_size=1 -> 2 bytes
                  | (1u << 20)    // pad_enable
                  | (3u << 22)    // pad_interval: 16 DWORDs
                  | (3u << 25));  // pad_amount: 4 DWORDs
    g1[1] = (int)((unsigned)(K & 0xFFFF) << 16);          // tensor_dim0 lo
    g1[2] = (int)(((unsigned)K >> 16) & 0xFFFFu) |
            (int)((unsigned)(M & 0xFFFF) << 16);          // dim0 hi | dim1 lo
    g1[3] = (int)((((unsigned)M >> 16) & 0xFFFFu) | (32u << 16));  // tile_dim0
    g1[4] = 128;  // tile_dim1=128 (tile_dim2=0)
    g1[5] = K;    // tensor_dim0_stride (low 32)
    g1[6] = 0;
    g1[7] = 0;
    v4i_t gz4 = {0, 0, 0, 0};
    v8i_t gz8 = {0, 0, 0, 0, 0, 0, 0, 0};
    __builtin_amdgcn_tensor_load_to_lds(g0, g1, gz4, gz4, gz8, 0);
  };
#else
  auto manual_a = [&](int kt, unsigned short* As) {
#pragma unroll
    for (int c = tid; c < 512; c += 256) {
      int r  = c >> 2;
      int cp = (c & 3) << 3;
      uint4 d = *reinterpret_cast<const uint4*>(
          &A[(size_t)(row0 + r) * K + kt + cp]);
      *reinterpret_cast<uint4*>(&As[r * LDT + cp]) = d;
    }
  };
#endif
  // B tile 32x128: each thread owns chunks tid and tid+256 (8 halves along n).
  auto loadB = [&](int kt, uint4& d0, uint4& d1) {
    int k0c = tid >> 4, n80 = (tid & 15) << 3;
    int c1 = tid + 256;
    int k1c = c1 >> 4, n81 = (c1 & 15) << 3;
    d0 = *reinterpret_cast<const uint4*>(&B[(size_t)(kt + k0c) * N + col0 + n80]);
    d1 = *reinterpret_cast<const uint4*>(&B[(size_t)(kt + k1c) * N + col0 + n81]);
  };
  auto storeB = [&](unsigned short* Bst, const uint4& d0, const uint4& d1) {
    {
      int k = tid >> 4, n8 = (tid & 15) << 3;
      const unsigned short* ds = reinterpret_cast<const unsigned short*>(&d0);
#pragma unroll
      for (int i = 0; i < 8; ++i) Bst[(n8 + i) * LDT + k] = ds[i];
    }
    {
      int c = tid + 256;
      int k = c >> 4, n8 = (c & 15) << 3;
      const unsigned short* ds = reinterpret_cast<const unsigned short*>(&d1);
#pragma unroll
      for (int i = 0; i < 8; ++i) Bst[(n8 + i) * LDT + k] = ds[i];
    }
  };

  v8f acc[2][4];
#pragma unroll
  for (int mi = 0; mi < 2; ++mi)
#pragma unroll
    for (int ni = 0; ni < 4; ++ni) acc[mi][ni] = v8f_zero();

  // ---- prologue: stage tile 0 ----------------------------------------
  uint4 b0, b1;
#if USE_TDM
  if (wave == 0) tdm_issue(0, 0u);
#else
  manual_a(0, smem);
#endif
  loadB(0, b0, b1);
  storeB(smem + 2 * TILE, b0, b1);
#if USE_TDM
  if (wave == 0) __builtin_amdgcn_s_wait_tensorcnt(0);
#endif
  __syncthreads();

  // ---- pipelined main loop --------------------------------------------
  for (int kt = 0; kt < K; kt += 32) {
    const int  cur      = (kt >> 5) & 1;
    const int  nxt      = cur ^ 1;
    const bool has_next = (kt + 32) < K;

    if (has_next) {
#if USE_TDM
      if (wave == 0) tdm_issue(kt + 32, (unsigned)nxt * TILE_B);  // async
#else
      manual_a(kt + 32, smem + nxt * TILE);
#endif
      loadB(kt + 32, b0, b1);  // global loads overlap compute below
    }

    const unsigned short* As = smem + cur * TILE;
    const unsigned short* Bs = smem + 2 * TILE + cur * TILE;
    // A fragments: lanes 0-15 -> K{0..7,16..23}; lanes 16-31 -> K{8..15,24..31}
    FragAB aF[2];
#pragma unroll
    for (int mi = 0; mi < 2; ++mi) {
      int r   = wm * 32 + mi * 16 + l16;
      int off = lhalf * 8;
      aF[mi].q[0] = *reinterpret_cast<const uint4*>(&As[r * LDT + off]);
      aF[mi].q[1] = *reinterpret_cast<const uint4*>(&As[r * LDT + 16 + off]);
    }
    // B fragments: lanes 0-15 -> K{0..7,16..23}; lanes 16-31 -> K{8..15,24..31}
    FragAB bF[4];
#pragma unroll
    for (int ni = 0; ni < 4; ++ni) {
      int n   = wn * 64 + ni * 16 + l16;
      int off = lhalf * 8;
      bF[ni].q[0] = *reinterpret_cast<const uint4*>(&Bs[n * LDT + off]);
      bF[ni].q[1] = *reinterpret_cast<const uint4*>(&Bs[n * LDT + off + 16]);
    }
#pragma unroll
    for (int mi = 0; mi < 2; ++mi)
#pragma unroll
      for (int ni = 0; ni < 4; ++ni)
        acc[mi][ni] = __builtin_amdgcn_wmma_f32_16x16x32_bf16(
            false, aF[mi].v, false, bF[ni].v, (short)0, acc[mi][ni], false,
            false);

    if (has_next) storeB(smem + 2 * TILE + nxt * TILE, b0, b1);
#if USE_TDM
    if (wave == 0) __builtin_amdgcn_s_wait_tensorcnt(0);  // drain next A tile
#endif
    __syncthreads();
  }

  // Epilogue: C layout -> lane l16 = N, vgpr r -> M = r + 8*lhalf.
#pragma unroll
  for (int mi = 0; mi < 2; ++mi)
#pragma unroll
    for (int ni = 0; ni < 4; ++ni)
#pragma unroll
      for (int r = 0; r < 8; ++r) {
        int m = row0 + wm * 32 + mi * 16 + r + 8 * lhalf;
        int n = col0 + wn * 64 + ni * 16 + l16;
        float vv = acc[mi][ni][r];
        if constexpr (OUT_F32)
          reinterpret_cast<float*>(Cout)[(size_t)m * N + n] = vv;
        else
          reinterpret_cast<unsigned short*>(Cout)[(size_t)m * N + n] = f2bf(vv);
      }
}

// --------------------------------- RoPE ------------------------------------
__global__ void rope_kernel(unsigned short* __restrict__ Q,
                            unsigned short* __restrict__ Kb,
                            const float* __restrict__ cosp,
                            const float* __restrict__ sinp, int rows, int T) {
  int i = blockIdx.x * blockDim.x + threadIdx.x;
  if (i >= rows * 16 * 32) return;
  int d   = i & 31;
  int h   = (i >> 5) & 15;
  int row = i >> 9;
  int t   = row & (T - 1);  // row % T, T power of 2
  float c0 = cosp[t * 64 + d],      s0 = sinp[t * 64 + d];
  float c1 = cosp[t * 64 + d + 32], s1 = sinp[t * 64 + d + 32];
  size_t base = (size_t)row * 1024 + h * 64 + d;
  {
    float q0 = bf2f(Q[base]), q1 = bf2f(Q[base + 32]);
    Q[base]      = f2bf(q0 * c0 - q1 * s0);
    Q[base + 32] = f2bf(q1 * c1 + q0 * s1);
  }
  {
    float k0 = bf2f(Kb[base]), k1 = bf2f(Kb[base + 32]);
    Kb[base]      = f2bf(k0 * c0 - k1 * s0);
    Kb[base + 32] = f2bf(k1 * c1 + k0 * s1);
  }
}

// ----------------------------- Flash attention -----------------------------
// Block = (b,h, 128 query rows), 8 waves, each wave owns 16 query rows.
// Key tiles of 32; online softmax in fp32; S and PV via bf16 WMMA.
// V tiles double-buffered: next tile's global loads overlap compute.
__global__ __launch_bounds__(256) void flash_attn_kernel(
    const unsigned short* __restrict__ Q, const unsigned short* __restrict__ Kb,
    const unsigned short* __restrict__ Vb, unsigned short* __restrict__ Ob,
    int T, int H) {
  constexpr int LDV = 40;
  __shared__ __align__(16) unsigned short Vs[2][64 * LDV];  // V^T [dh][key] x2
  __shared__ __align__(16) unsigned short Ps[8][16 * LDV];  // per-wave P 16x32

  const int tid   = threadIdx.x;
  const int wave  = tid >> 5;
  const int lane  = tid & 31;
  const int l16   = lane & 15;
  const int lhalf = lane >> 4;

  const int t0 = blockIdx.x * 128;
  const int bh = blockIdx.y;
  const int b  = bh / H, h = bh % H;
  const size_t rowbase = (size_t)b * T;
  const int D  = H * 64;
  const int q0 = t0 + wave * 16;

  auto loadV = [&](int k0) {
    int key = tid >> 3;
    int d8  = (tid & 7) << 3;
    return *reinterpret_cast<const uint4*>(
        &Vb[(rowbase + k0 + key) * D + h * 64 + d8]);
  };
  auto scatterV = [&](unsigned short* vs, const uint4& dv) {
    int key = tid >> 3;
    int d8  = (tid & 7) << 3;
    const unsigned short* ds = reinterpret_cast<const unsigned short*>(&dv);
#pragma unroll
    for (int i2 = 0; i2 < 8; ++i2) vs[(d8 + i2) * LDV + key] = ds[i2];
  };

  // Q fragments (A layout), Dh split into K-chunks 0..31 / 32..63.
  FragAB qF[2];
  {
    size_t qrow = (rowbase + q0 + l16) * D + h * 64;
    int off = lhalf * 8;
#pragma unroll
    for (int kc = 0; kc < 2; ++kc) {
      qF[kc].q[0] = *reinterpret_cast<const uint4*>(&Q[qrow + kc * 32 + off]);
      qF[kc].q[1] =
          *reinterpret_cast<const uint4*>(&Q[qrow + kc * 32 + 16 + off]);
    }
  }

  float mi[8], li[8];
  v8f   oacc[4];
#pragma unroll
  for (int r = 0; r < 8; ++r) { mi[r] = -3.0e38f; li[r] = 0.f; }
#pragma unroll
  for (int ni = 0; ni < 4; ++ni) oacc[ni] = v8f_zero();

  const int qmax = q0 + 15;
  const int nkt  = (t0 + 128) / 32;  // key tiles covering 0 .. t0+127

  // Prologue: stage V tile 0.
  {
    uint4 dv = loadV(0);
    scatterV(&Vs[0][0], dv);
  }
  __syncthreads();

  for (int it = 0; it < nkt; ++it) {
    const int  k0       = it * 32;
    const int  cur      = it & 1;
    const int  nxt      = cur ^ 1;
    const bool has_next = (it + 1) < nkt;

    uint4 dvn;
    if (has_next) dvn = loadV(k0 + 32);  // overlap with compute below

    if (k0 <= qmax) {  // wave-uniform: EXEC stays all-ones inside
      // S = Q K^T : two 16x16 n-tiles of keys; K rows are B-fragments as-is.
      v8f sac[2];
#pragma unroll
      for (int jn = 0; jn < 2; ++jn) {
        sac[jn] = v8f_zero();
#pragma unroll
        for (int kc = 0; kc < 2; ++kc) {
          FragAB kf;
          size_t krow = (rowbase + k0 + jn * 16 + l16) * D + h * 64 +
                        kc * 32 + lhalf * 8;
          kf.q[0] = *reinterpret_cast<const uint4*>(&Kb[krow]);
          kf.q[1] = *reinterpret_cast<const uint4*>(&Kb[krow + 16]);
          sac[jn] = __builtin_amdgcn_wmma_f32_16x16x32_bf16(
              false, qF[kc].v, false, kf.v, (short)0, sac[jn], false, false);
        }
      }
      // Online softmax; row m = r + 8*lhalf, key col = l16 (+16 for tile 1).
      float scale_r[8];
#pragma unroll
      for (int r = 0; r < 8; ++r) {
        const int m  = r + 8 * lhalf;
        const int qg = q0 + m;
        float s0v = sac[0][r] * 0.125f;  // 1/sqrt(64)
        float s1v = sac[1][r] * 0.125f;
        if (k0 + l16 > qg)      s0v = -1.0e30f;
        if (k0 + 16 + l16 > qg) s1v = -1.0e30f;
        float mx   = row16_max(fmaxf(s0v, s1v));  // DPP row_ror reduction
        float mnew = fmaxf(mi[r], mx);
        float sc   = __expf(mi[r] - mnew);
        float p0   = __expf(s0v - mnew);
        float p1   = __expf(s1v - mnew);
        float rs   = row16_sum(p0 + p1);          // DPP row_ror reduction
        li[r] = li[r] * sc + rs;
        mi[r] = mnew;
        scale_r[r] = sc;
        Ps[wave][m * LDV + l16]      = f2bf(p0);
        Ps[wave][m * LDV + 16 + l16] = f2bf(p1);
      }
      // Reload P (16x32) as an A fragment (per-wave LDS; same-wave ordering).
      FragAB pF;
      {
        int off = lhalf * 8;
        const unsigned short* pr = &Ps[wave][l16 * LDV];
        pF.q[0] = *reinterpret_cast<const uint4*>(&pr[off]);
        pF.q[1] = *reinterpret_cast<const uint4*>(&pr[16 + off]);
      }
      // O = O*diag(scale) + P V
#pragma unroll
      for (int ni = 0; ni < 4; ++ni) {
#pragma unroll
        for (int r = 0; r < 8; ++r) oacc[ni][r] *= scale_r[r];
        FragAB vf;
        const unsigned short* vr = &Vs[cur][(ni * 16 + l16) * LDV + lhalf * 8];
        vf.q[0] = *reinterpret_cast<const uint4*>(&vr[0]);
        vf.q[1] = *reinterpret_cast<const uint4*>(&vr[16]);
        oacc[ni] = __builtin_amdgcn_wmma_f32_16x16x32_bf16(
            false, pF.v, false, vf.v, (short)0, oacc[ni], false, false);
      }
    }

    if (has_next) scatterV(&Vs[nxt][0], dvn);
    __syncthreads();
  }

  // Normalize and store attention output (bf16, layout [B*T][H*64]).
  float inv[8];
#pragma unroll
  for (int r = 0; r < 8; ++r) inv[r] = 1.0f / li[r];
#pragma unroll
  for (int ni = 0; ni < 4; ++ni)
#pragma unroll
    for (int r = 0; r < 8; ++r) {
      int m = r + 8 * lhalf;
      Ob[(rowbase + q0 + m) * D + h * 64 + ni * 16 + l16] =
          f2bf(oacc[ni][r] * inv[r]);
    }
}

// --------------------------------- launch ----------------------------------
extern "C" void kernel_launch(void* const* d_in, const int* in_sizes, int n_in,
                              void* d_out, int out_size, void* d_ws,
                              size_t ws_size, hipStream_t stream) {
  const float* x    = (const float*)d_in[0];
  const float* cosp = (const float*)d_in[1];
  const float* sinp = (const float*)d_in[2];
  const float* Wq   = (const float*)d_in[3];
  const float* Wk   = (const float*)d_in[4];
  const float* Wv   = (const float*)d_in[5];
  const float* Wo   = (const float*)d_in[6];
  float* out = (float*)d_out;

  const int Bb = 4, T = 2048, D = 1024, H = 16;
  const int M = Bb * T;  // 8192

  unsigned short* ws  = (unsigned short*)d_ws;
  unsigned short* Xb  = ws;                       // M*D  (reused as attn out)
  unsigned short* Wqb = Xb + (size_t)M * D;       // D*D
  unsigned short* Wkb = Wqb + (size_t)D * D;
  unsigned short* Wvb = Wkb + (size_t)D * D;
  unsigned short* Wob = Wvb + (size_t)D * D;
  unsigned short* Qb  = Wob + (size_t)D * D;      // M*D
  unsigned short* Kb  = Qb + (size_t)M * D;       // M*D
  unsigned short* Vb  = Kb + (size_t)M * D;       // M*D
  unsigned short* Ab  = Xb;                       // attn out reuses Xb

  // 1) fp32 -> bf16 casts.
  {
    int nx = M * D;
    cvt_f32_bf16<<<(nx + 255) / 256, 256, 0, stream>>>(x, Xb, nx);
    int nw = D * D;
    cvt_f32_bf16<<<(nw + 255) / 256, 256, 0, stream>>>(Wq, Wqb, nw);
    cvt_f32_bf16<<<(nw + 255) / 256, 256, 0, stream>>>(Wk, Wkb, nw);
    cvt_f32_bf16<<<(nw + 255) / 256, 256, 0, stream>>>(Wv, Wvb, nw);
    cvt_f32_bf16<<<(nw + 255) / 256, 256, 0, stream>>>(Wo, Wob, nw);
  }

  const size_t gemm_lds = 4u * 128u * 40u * sizeof(unsigned short);  // 40960 B

  // 2) QKV projections (bf16 WMMA, bf16 out).
  dim3 gg(D / 128, M / 128);  // (8, 64)
  gemm_bf16_wmma<false><<<gg, 256, gemm_lds, stream>>>(Xb, Wqb, Qb, M, D, D);
  gemm_bf16_wmma<false><<<gg, 256, gemm_lds, stream>>>(Xb, Wkb, Kb, M, D, D);
  gemm_bf16_wmma<false><<<gg, 256, gemm_lds, stream>>>(Xb, Wvb, Vb, M, D, D);

  // 3) RoPE on Q and K.
  {
    int total = M * H * 32;
    rope_kernel<<<(total + 255) / 256, 256, 0, stream>>>(Qb, Kb, cosp, sinp, M,
                                                         T);
  }

  // 4) Flash attention -> Ab (bf16 [B*T][H*64]).
  {
    dim3 fg(T / 128, Bb * H);  // (16, 64)
    flash_attn_kernel<<<fg, 256, 0, stream>>>(Qb, Kb, Vb, Ab, T, H);
  }

  // 5) Output projection (fp32 out).
  gemm_bf16_wmma<true><<<gg, 256, gemm_lds, stream>>>(Ab, Wob, out, M, D, D);
}
